// GraphAttentionEmbedding_70944269795721
// MI455X (gfx1250) — compile-verified, matrix operand order checked by reading, not run on hardware
//
#include <hip/hip_runtime.h>
#include <math.h>

// ---------------- WMMA types ----------------
typedef __attribute__((ext_vector_type(16))) __bf16 v16bf;
typedef __attribute__((ext_vector_type(8)))  float  v8f;

static __device__ __forceinline__ __bf16 f2bf(float f) {
  unsigned u = __builtin_bit_cast(unsigned, f);
  u += 0x7FFFu + ((u >> 16) & 1u);            // round-to-nearest-even
  unsigned short h = (unsigned short)(u >> 16);
  return __builtin_bit_cast(__bf16, h);
}

struct Bytes32 { uint4 a, b; };
static __device__ __forceinline__ void store32B(unsigned short* p, v16bf v) {
  Bytes32 t = __builtin_bit_cast(Bytes32, v);
  *reinterpret_cast<uint4*>(p)     = t.a;
  *reinterpret_cast<uint4*>(p + 8) = t.b;
}

static __device__ __forceinline__ void atomAddF(float* p, float v) {
  __hip_atomic_fetch_add(p, v, __ATOMIC_RELAXED, __HIP_MEMORY_SCOPE_AGENT);
}

// float atomic max via ordered-int trick (global_atomic_max_i32 / min_u32)
static __device__ __forceinline__ void atomMaxF(float* addr, float val) {
  if (val >= 0.0f)
    atomicMax(reinterpret_cast<int*>(addr), __float_as_int(val));
  else
    atomicMin(reinterpret_cast<unsigned int*>(addr), __float_as_uint(val));
}

// =====================================================================
// Fragment packing. A WMMA bf16 fragment = 512 bf16 (32 lanes x 16 elems).
// Per ISA 16-bit A 16x32 layout: lane<16 -> row M=lane, K = {0..7,16..23};
// lane>=16 -> row M=lane-16, K = {8..15,24..31}. B mirrors with M->N.
// Storage: fragment f at Xp + f*512, lane chunk at +lane*16 (32B contig).
// =====================================================================

// ---- pack activations X (rows x KD, f32 row-major) into A fragments ----
template <int KD>
__global__ __launch_bounds__(128)
void pack_a(const float* __restrict__ X, unsigned short* __restrict__ Xp, int rows) {
  constexpr int KB = KD / 32;
  const int wid  = (int)((blockIdx.x * blockDim.x + threadIdx.x) >> 5);
  const int lane = threadIdx.x & 31;
  const int rtiles = (rows + 15) / 16;
  const int rt = wid / KB, kbt = wid % KB;
  if (rt >= rtiles) return;
  const int l15 = lane & 15, hs = lane >> 4;
  int row = rt * 16 + l15;
  if (row >= rows) row = rows - 1;           // clamp (duplicate pad rows)
  const int kbase = kbt * 32 + hs * 8;
  const float* xp = X + (size_t)row * KD + kbase;
  const float4 a0 = *reinterpret_cast<const float4*>(xp + 0);
  const float4 a1 = *reinterpret_cast<const float4*>(xp + 4);
  const float4 a2 = *reinterpret_cast<const float4*>(xp + 16);
  const float4 a3 = *reinterpret_cast<const float4*>(xp + 20);
  v16bf av;
  av[0]=f2bf(a0.x); av[1]=f2bf(a0.y); av[2]=f2bf(a0.z); av[3]=f2bf(a0.w);
  av[4]=f2bf(a1.x); av[5]=f2bf(a1.y); av[6]=f2bf(a1.z); av[7]=f2bf(a1.w);
  av[8]=f2bf(a2.x); av[9]=f2bf(a2.y); av[10]=f2bf(a2.z); av[11]=f2bf(a2.w);
  av[12]=f2bf(a3.x); av[13]=f2bf(a3.y); av[14]=f2bf(a3.z); av[15]=f2bf(a3.w);
  store32B(Xp + (size_t)wid * 512 + lane * 16, av);
}

// ---- pack weights W (KD x 64, f32 row-major) into B fragments ----
// fragment index = kbt*4 + nt  (nt = 16-col tile)
template <int KD>
__global__ __launch_bounds__(128)
void pack_b(const float* __restrict__ W, unsigned short* __restrict__ Wp) {
  constexpr int F = (KD / 32) * 4;
  const int wid  = (int)((blockIdx.x * blockDim.x + threadIdx.x) >> 5);
  const int lane = threadIdx.x & 31;
  if (wid >= F) return;
  const int kbt = wid >> 2, nt = wid & 3;
  const int l15 = lane & 15, hs = lane >> 4;
  const int kbase = kbt * 32 + hs * 8;
  const int n = nt * 16 + l15;
  v16bf bv;
#pragma unroll
  for (int e = 0; e < 16; ++e) {
    const int kk = (e < 8) ? e : (e + 8);
    bv[e] = f2bf(W[(size_t)(kbase + kk) * 64 + n]);
  }
  store32B(Wp + (size_t)wid * 512 + lane * 16, bv);
}

// =====================================================================
// Packed GEMM: O = Xp(rows x KD, fragments) @ Wp(KD x 64, fragments) + b
// blockDim=128 (4 waves), wave = 16-col tile; grid.x = row tiles,
// grid.y selects one of up to 4 (W,b,O) sets. Inner loop: 2x32B load + wmma.
// =====================================================================
struct GemmArgs {
  const unsigned short* Xp;
  const unsigned short* Wp[4];
  const float*          b[4];
  float*                O[4];
  int                   rows;
};

template <int KD>
__global__ __launch_bounds__(128)
void wmma_gemm(GemmArgs g) {
  constexpr int KB = KD / 32;
  const int mat  = blockIdx.y;
  const unsigned short* __restrict__ Wp = g.Wp[mat];
  const float* __restrict__ bias = g.b[mat];
  float* __restrict__ O = g.O[mat];

  const int lane = threadIdx.x & 31;
  const int wv   = threadIdx.x >> 5;       // col tile
  const int rt   = blockIdx.x;
  const int l15  = lane & 15;
  const int hs   = lane >> 4;

  const unsigned short* ap = g.Xp + ((size_t)rt * KB) * 512 + lane * 16;
  const unsigned short* bp = Wp + (size_t)wv * 512 + lane * 16;

  v8f acc = {0.f,0.f,0.f,0.f,0.f,0.f,0.f,0.f};
#pragma unroll
  for (int kbt = 0; kbt < KB; ++kbt) {
    const v16bf av = *reinterpret_cast<const v16bf*>(ap + (size_t)kbt * 512);
    const v16bf bv = *reinterpret_cast<const v16bf*>(bp + (size_t)kbt * 2048);
    acc = __builtin_amdgcn_wmma_f32_16x16x32_bf16(
        false, av, false, bv, (short)0, acc, false, false);
  }

  const int c = wv * 16 + l15;
  const float badd = bias ? bias[c] : 0.0f;
  float* op = O + ((size_t)rt * 16 + (hs << 3)) * 64 + c;  // row rt*16 + hs*8
  if ((rt + 1) * 16 <= g.rows) {
    // full tile (always the case for multiple-of-16 row counts): straight-line
#pragma unroll
    for (int v = 0; v < 8; ++v) op[(size_t)v * 64] = acc[v] + badd;
  } else {
#pragma unroll
    for (int v = 0; v < 8; ++v) {
      const int r = rt * 16 + v + (hs << 3);
      if (r < g.rows) op[(size_t)v * 64] = acc[v] + badd;
    }
  }
}

// ---------------- edge pass 1: alpha = scale * <q[dst], k[src]+ee>, atomic segment max ----------------
template <int Hh, int Cc>
__global__ void edge_alpha(const int* __restrict__ src, const int* __restrict__ dst,
                           const float* __restrict__ Q, const float* __restrict__ K,
                           const float* __restrict__ EE, float* __restrict__ EXb,
                           float* __restrict__ AMAX, int E_, float scale) {
  const int e = blockIdx.x * blockDim.x + threadIdx.x;
  if (e >= E_) return;
  const int s = src[e], d = dst[e];
  const float4* qd = reinterpret_cast<const float4*>(Q  + (size_t)d * 64);
  const float4* ks = reinterpret_cast<const float4*>(K  + (size_t)s * 64);
  const float4* ee = reinterpret_cast<const float4*>(EE + (size_t)e * 64);
#pragma unroll
  for (int h = 0; h < Hh; ++h) {
    float acc = 0.f;
#pragma unroll
    for (int i = 0; i < Cc / 4; ++i) {
      const float4 q4 = qd[h * (Cc / 4) + i];
      const float4 k4 = ks[h * (Cc / 4) + i];
      const float4 e4 = ee[h * (Cc / 4) + i];
      acc += q4.x * (k4.x + e4.x) + q4.y * (k4.y + e4.y)
           + q4.z * (k4.z + e4.z) + q4.w * (k4.w + e4.w);
    }
    const float alpha = acc * scale;
    EXb[(size_t)e * Hh + h] = alpha;
    atomMaxF(&AMAX[(size_t)d * Hh + h], alpha);
  }
}

// ---------------- edge pass 2: ex = exp(alpha - amax[dst]); denom += ex ----------------
template <int Hh>
__global__ void edge_exp(const int* __restrict__ dst, float* __restrict__ EXb,
                         const float* __restrict__ AMAX, float* __restrict__ DEN, int E_) {
  const int e = blockIdx.x * blockDim.x + threadIdx.x;
  if (e >= E_) return;
  const int d = dst[e];
#pragma unroll
  for (int h = 0; h < Hh; ++h) {
    const float a  = EXb[(size_t)e * Hh + h];
    const float ex = __expf(a - AMAX[(size_t)d * Hh + h]);
    EXb[(size_t)e * Hh + h] = ex;
    atomAddF(&DEN[(size_t)d * Hh + h], ex);
  }
}

// ---------------- edge pass 3: out[dst] += (v[src]+ee) * ex/denom  (thread per edge x 4ch) ----------------
template <int Hh, int Cc>
__global__ void edge_scatter(const int* __restrict__ src, const int* __restrict__ dst,
                             const float* __restrict__ V, const float* __restrict__ EE,
                             const float* __restrict__ EXb, const float* __restrict__ DEN,
                             float* __restrict__ Out, int E_) {
  const size_t t = blockIdx.x * (size_t)blockDim.x + threadIdx.x;
  const size_t e = t >> 4;                 // 16 groups of 4 channels = 64 channels
  if (e >= (size_t)E_) return;
  const int j0 = (int)(t & 15) * 4;        // channel 0..63
  const int h  = j0 / Cc;
  const int s = src[e], d = dst[e];
  const float attn = EXb[e * Hh + h] / (DEN[(size_t)d * Hh + h] + 1e-16f);
  const float4 v4 = *reinterpret_cast<const float4*>(V  + (size_t)s * 64 + j0);
  const float4 e4 = *reinterpret_cast<const float4*>(EE + e * 64 + j0);
  float* o = Out + (size_t)d * 64 + j0;
  atomAddF(o + 0, (v4.x + e4.x) * attn);
  atomAddF(o + 1, (v4.y + e4.y) * attn);
  atomAddF(o + 2, (v4.z + e4.z) * attn);
  atomAddF(o + 3, (v4.w + e4.w) * attn);
}

// ---------------- utility kernels ----------------
__global__ void fill_f32(float* p, size_t n, float v) {
  const size_t i = blockIdx.x * (size_t)blockDim.x + threadIdx.x;
  if (i < n) p[i] = v;
}
__global__ void relu_f32(float* p, size_t n) {
  const size_t i = blockIdx.x * (size_t)blockDim.x + threadIdx.x;
  if (i < n) p[i] = fmaxf(p[i], 0.0f);
}

// ---------------- host orchestration ----------------
extern "C" void kernel_launch(void* const* d_in, const int* in_sizes, int n_in,
                              void* d_out, int out_size, void* d_ws, size_t ws_size,
                              hipStream_t stream) {
  const float* x   = (const float*)d_in[0];
  const int*   ei  = (const int*)  d_in[1];
  const float* ef  = (const float*)d_in[2];
  const float* Wq1 = (const float*)d_in[3];  const float* bq1 = (const float*)d_in[4];
  const float* Wk1 = (const float*)d_in[5];  const float* bk1 = (const float*)d_in[6];
  const float* Wv1 = (const float*)d_in[7];  const float* bv1 = (const float*)d_in[8];
  const float* We1 = (const float*)d_in[9];
  const float* Ws1 = (const float*)d_in[10]; const float* bs1 = (const float*)d_in[11];
  const float* Wq2 = (const float*)d_in[12]; const float* bq2 = (const float*)d_in[13];
  const float* Wk2 = (const float*)d_in[14]; const float* bk2 = (const float*)d_in[15];
  const float* Wv2 = (const float*)d_in[16]; const float* bv2 = (const float*)d_in[17];
  const float* We2 = (const float*)d_in[18];
  const float* Ws2 = (const float*)d_in[19]; const float* bs2 = (const float*)d_in[20];
  (void)n_in; (void)out_size; (void)ws_size;

  const int Nn = in_sizes[0] / 128;      // 100000
  const int Ee = in_sizes[2] / 32;       // 1000000
  const int* srcI = ei;
  const int* dstI = ei + Ee;
  const int NT = (Nn + 15) / 16;         // node row tiles
  const int ET = (Ee + 15) / 16;         // edge row tiles

  // ---- workspace layout (256B-aligned regions) ----
  size_t off = 0;
  auto alloc = [&](size_t bytes) -> void* {
    void* p = (char*)d_ws + off;
    off = (off + bytes + 255) & ~(size_t)255;
    return p;
  };
  const size_t n64 = (size_t)Nn * 64;
  float* Q   = (float*)alloc(n64 * 4);
  float* Kf  = (float*)alloc(n64 * 4);
  float* V   = (float*)alloc(n64 * 4);
  float* Hb  = (float*)alloc(n64 * 4);
  float* AM  = (float*)alloc((size_t)Nn * 2 * 4);
  float* DN  = (float*)alloc((size_t)Nn * 2 * 4);
  float* EXb = (float*)alloc((size_t)Ee * 2 * 4);
  float* EE  = (float*)alloc((size_t)Ee * 64 * 4);
  typedef unsigned short u16;
  u16* Xp  = (u16*)alloc((size_t)NT * 4 * 512 * 2 * (128 / 32)); // K=128 fragments
  u16* EFp = (u16*)alloc((size_t)ET * 512 * 2 * (32 / 32));      // K=32
  u16* Hp  = (u16*)alloc((size_t)NT * 512 * 2 * (64 / 32) * 2);  // K=64 (x2 slack)
  u16* WP[10];
  for (int i = 0; i < 10; ++i) WP[i] = (u16*)alloc(8192 * 2);    // max (128/32)*4*512

  float* OUT = (float*)d_out;
  const dim3 blk256(256), blk128(128);
  const unsigned gN2  = (unsigned)(((size_t)Nn * 2 + 255) / 256);
  const unsigned gE   = (unsigned)((Ee + 255) / 256);
  const unsigned gE16 = (unsigned)(((size_t)Ee * 16 + 255) / 256);
  const unsigned gN64 = (unsigned)((n64 + 255) / 256);
  const float NEGINF = -__builtin_huge_valf();
  auto wgrid = [](size_t waves) { return (unsigned)((waves + 3) / 4); };

  // ---- pack all weights (once) and activations ----
  pack_b<128><<<wgrid(16), blk128, 0, stream>>>(Wq1, WP[0]);
  pack_b<128><<<wgrid(16), blk128, 0, stream>>>(Wk1, WP[1]);
  pack_b<128><<<wgrid(16), blk128, 0, stream>>>(Wv1, WP[2]);
  pack_b<128><<<wgrid(16), blk128, 0, stream>>>(Ws1, WP[3]);
  pack_b<32> <<<wgrid(4),  blk128, 0, stream>>>(We1, WP[4]);
  pack_b<64> <<<wgrid(8),  blk128, 0, stream>>>(Wq2, WP[5]);
  pack_b<64> <<<wgrid(8),  blk128, 0, stream>>>(Wk2, WP[6]);
  pack_b<64> <<<wgrid(8),  blk128, 0, stream>>>(Wv2, WP[7]);
  pack_b<64> <<<wgrid(8),  blk128, 0, stream>>>(Ws2, WP[8]);
  pack_b<32> <<<wgrid(4),  blk128, 0, stream>>>(We2, WP[9]);
  pack_a<128><<<wgrid((size_t)NT * 4), blk128, 0, stream>>>(x,  Xp,  Nn);
  pack_a<32> <<<wgrid((size_t)ET),     blk128, 0, stream>>>(ef, EFp, Ee);

  // ================= layer 1 (H=2, C=32) =================
  fill_f32<<<gN2, blk256, 0, stream>>>(AM, (size_t)Nn * 2, NEGINF);
  fill_f32<<<gN2, blk256, 0, stream>>>(DN, (size_t)Nn * 2, 0.0f);

  GemmArgs p1{ Xp, {WP[0], WP[1], WP[2], WP[3]}, {bq1, bk1, bv1, bs1},
               {Q, Kf, V, Hb}, Nn };
  wmma_gemm<128><<<dim3((unsigned)NT, 4), blk128, 0, stream>>>(p1);

  GemmArgs pe1{ EFp, {WP[4], WP[4], WP[4], WP[4]},
                {nullptr, nullptr, nullptr, nullptr}, {EE, EE, EE, EE}, Ee };
  wmma_gemm<32><<<dim3((unsigned)ET, 1), blk128, 0, stream>>>(pe1);

  edge_alpha<2, 32><<<gE, blk256, 0, stream>>>(srcI, dstI, Q, Kf, EE, EXb, AM, Ee,
                                               0.1767766953f /* 1/sqrt(32) */);
  edge_exp<2><<<gE, blk256, 0, stream>>>(dstI, EXb, AM, DN, Ee);
  edge_scatter<2, 32><<<gE16, blk256, 0, stream>>>(srcI, dstI, V, EE, EXb, DN, Hb, Ee);
  relu_f32<<<gN64, blk256, 0, stream>>>(Hb, n64);

  // ================= layer 2 (H=1, C=64) =================
  fill_f32<<<gN2, blk256, 0, stream>>>(AM, (size_t)Nn, NEGINF);
  fill_f32<<<gN2, blk256, 0, stream>>>(DN, (size_t)Nn, 0.0f);

  pack_a<64><<<wgrid((size_t)NT * 2), blk128, 0, stream>>>(Hb, Hp, Nn);

  GemmArgs p2{ Hp, {WP[5], WP[6], WP[7], WP[8]}, {bq2, bk2, bv2, bs2},
               {Q, Kf, V, OUT}, Nn };
  wmma_gemm<64><<<dim3((unsigned)NT, 4), blk128, 0, stream>>>(p2);

  GemmArgs pe2{ EFp, {WP[9], WP[9], WP[9], WP[9]},
                {nullptr, nullptr, nullptr, nullptr}, {EE, EE, EE, EE}, Ee };
  wmma_gemm<32><<<dim3((unsigned)ET, 1), blk128, 0, stream>>>(pe2);

  edge_alpha<1, 64><<<gE, blk256, 0, stream>>>(srcI, dstI, Q, Kf, EE, EXb, AM, Ee,
                                               0.125f /* 1/sqrt(64) */);
  edge_exp<1><<<gE, blk256, 0, stream>>>(dstI, EXb, AM, DN, Ee);
  edge_scatter<1, 64><<<gE16, blk256, 0, stream>>>(srcI, dstI, V, EE, EXb, DN, OUT, Ee);
}
